// Qwen2VLVisionEncoder_58772332478603
// MI455X (gfx1250) — matrix operand order repaired and checked by reading, not compiled, not run on hardware
//
#include <hip/hip_runtime.h>

// ---------------------------------------------------------------------------
// Qwen2-VL vision encoder for MI455X (gfx1250, wave32, WMMA + TDM)
// All GEMMs: v_wmma_f32_16x16x32_bf16, fp32 accum. A-tiles staged into LDS
// by the Tensor Data Mover (tensor_load_to_lds, double-buffered, TENSORcnt),
// B-tiles converted fp32->bf16 on the fly (HBM traffic stays at fp32 floor,
// 192MB L2 holds weights across M-tiles). Attention: fused flash-style,
// 1 wave per (head, 16-query tile).
// ---------------------------------------------------------------------------

#define SEQ    1024
#define EMBED  1280
#define HEADS  16
#define HD     80
#define HDP    96          // head dim padded to multiple of 32 for WMMA K
#define MLPDIM 5120
#define HIDDEN 3584
#define DEPTH  8
#define PATCH  1176

typedef __attribute__((ext_vector_type(16))) __bf16 v16bf;
typedef __attribute__((ext_vector_type(8)))  float  v8f;
typedef __attribute__((ext_vector_type(4)))  unsigned int u32x4;
typedef __attribute__((ext_vector_type(8)))  int i32x8;
typedef __attribute__((ext_vector_type(4)))  int i32x4;

union Frag { unsigned int u[8]; v16bf v; };
union Acc  { float f[8];        v8f   v; };

__device__ __forceinline__ unsigned short f2bf(float f) {
    unsigned int u = __float_as_uint(f);
    unsigned int r = u + 0x7FFFu + ((u >> 16) & 1u);   // round-to-nearest-even
    return (unsigned short)(r >> 16);
}

// ---------------------------------------------------------------------------
// Tensor Data Mover: async copy of a 128x32 bf16 tile (strided 2D tensor)
// from global to LDS.  Descriptor per CDNA5 ISA ch.8 (D# groups 0/1).
// 5-arg builtin on ROCm7.2/clang-22, 6-arg on clang-23.
// ---------------------------------------------------------------------------
#if defined(__has_builtin)
#  if __has_builtin(__builtin_amdgcn_tensor_load_to_lds) && \
      __has_builtin(__builtin_amdgcn_s_wait_tensorcnt)
#    define HAVE_TDM 1
#  endif
#endif

#ifdef HAVE_TDM
__device__ __forceinline__ void tdm_load_tile_a(
    const unsigned short* A, int lda, int M, int K,
    int m0, int k0, unsigned lds_off)
{
    unsigned long long ga =
        (unsigned long long)(uintptr_t)(A + (size_t)m0 * lda + k0);
    unsigned td0 = (unsigned)(K - k0);     // elements valid along dim0 (OOB->0)
    unsigned td1 = (unsigned)(M - m0);     // rows valid along dim1
    unsigned long long st = (unsigned long long)lda;   // dim0 stride (elements)

    // D# group 0: count=1 | lds_addr | global_addr[56:0] | type=2 (image)
    u32x4 g0 = {
        1u,
        lds_off,
        (unsigned)(ga & 0xFFFFFFFFu),
        (unsigned)((ga >> 32) & 0x1FFFFFFu) | (2u << 30)
    };
    // D# group 1: data_size=2B, tensor_dim0/1, tile_dim0=32, tile_dim1=128,
    //             tensor_dim0_stride = lda
    i32x8 g1 = {
        (int)(1u << 16),                                        // data_size=1 (2B)
        (int)((td0 & 0xFFFFu) << 16),                           // tensor_dim0 lo
        (int)(((td0 >> 16) & 0xFFFFu) | ((td1 & 0xFFFFu) << 16)),
        (int)(((td1 >> 16) & 0xFFFFu) | (32u << 16)),           // tile_dim0=32
        (int)128,                                               // tile_dim1=128
        (int)(st & 0xFFFFFFFFull),                              // dim0 stride lo
        (int)((st >> 32) & 0xFFFFull),                          // stride hi
        0
    };
    i32x4 z4 = { 0, 0, 0, 0 };
#if __clang_major__ >= 23
    i32x8 z8 = { 0, 0, 0, 0, 0, 0, 0, 0 };
    __builtin_amdgcn_tensor_load_to_lds(g0, g1, z4, z4, z8, 0);
#else
    __builtin_amdgcn_tensor_load_to_lds(g0, g1, z4, z4, 0);
#endif
}
#endif

// ---------------------------------------------------------------------------
// Generic WMMA GEMM: C(MxN) = epi(A_bf16(MxK) * toBF16(B_f32(KxN)) + bias)
// EPI: 0 = none, 1 = +resid, 2 = x*sigmoid(1.702x), 3 = tanh-GELU
// A tile: TDM async double-buffered into LDS.  B tile: VALU-converted.
// ---------------------------------------------------------------------------
template<int EPI, bool OUTBF>
__global__ __launch_bounds__(256)
void gemm_wmma(const unsigned short* __restrict__ A, int lda,
               const float* __restrict__ B, int ldb,
               const float* __restrict__ bias,
               const float* __restrict__ resid,
               void* __restrict__ Cout, int ldc,
               int M, int N, int K)
{
    __shared__ unsigned short As[2][128][32];   // double-buffered, 16 KB
    __shared__ unsigned short Bs[128][32];      // [n][k] transposed stage, 8 KB

    const int t    = threadIdx.x;
    const int lane = t & 31;
    const int wave = t >> 5;
    const int half = lane >> 4;
    const int l15  = lane & 15;
    const int wm   = (wave >> 2) * 64;
    const int wn   = (wave &  3) * 32;
    const int m0   = blockIdx.y * 128;
    const int n0   = blockIdx.x * 128;

    Acc acc[4][2];
#pragma unroll
    for (int a = 0; a < 4; ++a)
#pragma unroll
        for (int b = 0; b < 2; ++b)
#pragma unroll
            for (int r = 0; r < 8; ++r) acc[a][b].f[r] = 0.f;

    const int ktiles = (K + 31) / 32;

#ifdef HAVE_TDM
    const bool tdmw = (wave == 0);
    if (tdmw)
        tdm_load_tile_a(A, lda, M, K, m0, 0,
                        (unsigned)(uintptr_t)&As[0][0][0]);
#endif

    for (int kt = 0; kt < ktiles; ++kt) {
        const int k0  = kt * 32;
        const int buf = kt & 1;

        // ---- stage B tile (32x128 f32 -> bf16, transposed to [n][k]) ----
#pragma unroll
        for (int i = 0; i < 16; ++i) {
            int e = i * 256 + t;
            int k = e >> 7;
            int n = e & 127;
            float bv = 0.f;
            if (k0 + k < K) bv = B[(size_t)(k0 + k) * ldb + n0 + n];
            Bs[n][k] = f2bf(bv);
        }

#ifdef HAVE_TDM
        // kick TDM for next A tile into the other buffer, then drain current
        if (tdmw && kt + 1 < ktiles)
            tdm_load_tile_a(A, lda, M, K, m0, k0 + 32,
                            (unsigned)(uintptr_t)&As[buf ^ 1][0][0]);
        if (tdmw) {
            if (kt + 1 < ktiles) __builtin_amdgcn_s_wait_tensorcnt(1);
            else                 __builtin_amdgcn_s_wait_tensorcnt(0);
        }
#else
        // ---- fallback: stage A tile with VALU (2048 dwords, 8/thread) ----
#pragma unroll
        for (int i = 0; i < 8; ++i) {
            int e   = t * 8 + i;
            int row = e >> 4;
            int kk  = (e & 15) * 2;
            int gk  = k0 + kk;
            unsigned int val = 0;
            if (gk + 1 < K) {
                val = *(const unsigned int*)(A + (size_t)(m0 + row) * lda + gk);
            } else if (gk < K) {
                val = (unsigned int)A[(size_t)(m0 + row) * lda + gk];
            }
            *(unsigned int*)&As[buf][row][kk] = val;
        }
#endif
        __syncthreads();

        // prefetch next B tile into GL2 (global_prefetch_b8)
        if (kt + 1 < ktiles && (k0 + 32) < K) {
            const float* p = &B[(size_t)(k0 + 32) * ldb + n0 + (t & 127)];
            __builtin_prefetch(p, 0, 1);
        }

        // ---- load fragments (ISA 7.12.2 layouts) ----
        Frag afr[4], bfr[2];
#pragma unroll
        for (int ms = 0; ms < 4; ++ms) {
            int m = wm + ms * 16 + l15;
#pragma unroll
            for (int j = 0; j < 8; ++j) {
                int kk = ((j >= 4) ? 16 : 0) + half * 8 + (j & 3) * 2;
                afr[ms].u[j] = *(const unsigned int*)&As[buf][m][kk];
            }
        }
#pragma unroll
        for (int ns = 0; ns < 2; ++ns) {
            int n = wn + ns * 16 + l15;
            int kb = half * 8;
#pragma unroll
            for (int j = 0; j < 8; ++j)
                bfr[ns].u[j] = *(const unsigned int*)&Bs[n][kb + ((j >= 4) ? 16 : 0) + (j & 3) * 2];
        }
#pragma unroll
        for (int ms = 0; ms < 4; ++ms)
#pragma unroll
            for (int ns = 0; ns < 2; ++ns)
                acc[ms][ns].v = __builtin_amdgcn_wmma_f32_16x16x32_bf16(
                    false, afr[ms].v, false, bfr[ns].v,
                    (short)0, acc[ms][ns].v, false, false);
        __syncthreads();
    }

    // ---- epilogue (C layout: VGPR r -> row r + half*8, col l15) ----
#pragma unroll
    for (int ms = 0; ms < 4; ++ms) {
        int mb = m0 + wm + ms * 16 + half * 8;
#pragma unroll
        for (int ns = 0; ns < 2; ++ns) {
            int n  = n0 + wn + ns * 16 + l15;
            float bb = bias ? bias[n] : 0.f;
#pragma unroll
            for (int r = 0; r < 8; ++r) {
                int m = mb + r;
                float v = acc[ms][ns].f[r] + bb;
                if constexpr (EPI == 1) {
                    v += resid[(size_t)m * ldc + n];
                } else if constexpr (EPI == 2) {
                    v = v / (1.f + expf(-1.702f * v));
                } else if constexpr (EPI == 3) {
                    float u3 = v * v * v;
                    v = 0.5f * v * (1.f + tanhf(0.7978845608f * (v + 0.044715f * u3)));
                }
                if constexpr (OUTBF)
                    ((unsigned short*)Cout)[(size_t)m * ldc + n] = f2bf(v);
                else
                    ((float*)Cout)[(size_t)m * ldc + n] = v;
            }
        }
    }
}

// ---------------------------------------------------------------------------
// LayerNorm (fp32 in, bf16 out) — one block per token
// ---------------------------------------------------------------------------
__global__ __launch_bounds__(256)
void layernorm_bf16(const float* __restrict__ x,
                    const float* __restrict__ g,
                    const float* __restrict__ b,
                    unsigned short* __restrict__ y, int N)
{
    const int row = blockIdx.x;
    const float* xr = x + (size_t)row * N;
    __shared__ float red[8];

    float s = 0.f;
    for (int i = threadIdx.x; i < N; i += 256) s += xr[i];
#pragma unroll
    for (int m = 16; m >= 1; m >>= 1) s += __shfl_xor(s, m, 32);
    if ((threadIdx.x & 31) == 0) red[threadIdx.x >> 5] = s;
    __syncthreads();
    float mean = 0.f;
#pragma unroll
    for (int i = 0; i < 8; ++i) mean += red[i];
    mean /= (float)N;
    __syncthreads();

    float v = 0.f;
    for (int i = threadIdx.x; i < N; i += 256) {
        float d = xr[i] - mean; v += d * d;
    }
#pragma unroll
    for (int m = 16; m >= 1; m >>= 1) v += __shfl_xor(v, m, 32);
    if ((threadIdx.x & 31) == 0) red[threadIdx.x >> 5] = v;
    __syncthreads();
    float var = 0.f;
#pragma unroll
    for (int i = 0; i < 8; ++i) var += red[i];
    var /= (float)N;
    float rstd = rsqrtf(var + 1e-6f);

    for (int i = threadIdx.x; i < N; i += 256)
        y[(size_t)row * N + i] = f2bf((xr[i] - mean) * rstd * g[i] + b[i]);
}

// ---------------------------------------------------------------------------
// RoPE + repack: qkv f32 (SEQ x 3*EMBED) ->
//   qr,kr : [H][SEQ][HDP] bf16 (RoPE applied, d=80..95 zero)
//   vt    : [H][HD][SEQ]  bf16 (transposed for WMMA B fragments)
// ---------------------------------------------------------------------------
__global__ __launch_bounds__(256)
void rope_pack(const float* __restrict__ qkv,
               unsigned short* __restrict__ qr,
               unsigned short* __restrict__ kr,
               unsigned short* __restrict__ vt)
{
    const int s = blockIdx.x;
    int a   = s >> 6;
    int rem = s & 63;
    int c   = rem >> 2;
    int b2  = (rem >> 1) & 1;
    int d2  = rem & 1;
    float hp = (float)(2 * a + b2);
    float wp = (float)(2 * c + d2);

    for (int idx = threadIdx.x; idx < HEADS * HD; idx += 256) {
        int head = idx / HD;
        int d    = idx % HD;
        int i40  = (d < 40) ? d : (d - 40);
        float pos = (i40 < 20) ? hp : wp;
        int ii    = (i40 < 20) ? i40 : (i40 - 20);
        float inv = powf(10000.f, -(float)ii / 20.f);
        float ang = pos * inv;
        float cs = cosf(ang), sn = sinf(ang);

        size_t base = (size_t)s * (3 * EMBED) + head * HD;
        float q = qkv[base + d];
        float k = qkv[base + EMBED + d];
        float v = qkv[base + 2 * EMBED + d];
        float qo, ko;
        if (d < 40) {
            qo = q * cs - qkv[base + d + 40] * sn;
            ko = k * cs - qkv[base + EMBED + d + 40] * sn;
        } else {
            qo = q * cs + qkv[base + d - 40] * sn;
            ko = k * cs + qkv[base + EMBED + d - 40] * sn;
        }
        size_t o = ((size_t)head * SEQ + s) * HDP + d;
        qr[o] = f2bf(qo);
        kr[o] = f2bf(ko);
        vt[((size_t)head * HD + d) * SEQ + s] = f2bf(v);
    }
    for (int idx = threadIdx.x; idx < HEADS * (HDP - HD); idx += 256) {
        int head = idx / (HDP - HD);
        int d    = HD + idx % (HDP - HD);
        size_t o = ((size_t)head * SEQ + s) * HDP + d;
        qr[o] = 0; kr[o] = 0;
    }
}

// ---------------------------------------------------------------------------
// Fused flash attention: one wave per (head, 16-query tile).
// ---------------------------------------------------------------------------
__global__ __launch_bounds__(32)
void attn_wmma(const unsigned short* __restrict__ qr,
               const unsigned short* __restrict__ kr,
               const unsigned short* __restrict__ vt,
               unsigned short* __restrict__ obf)
{
    const int head = blockIdx.x >> 6;
    const int q0   = (blockIdx.x & 63) * 16;
    const int lane = threadIdx.x & 31;
    const int half = lane >> 4;
    const int l15  = lane & 15;
    __shared__ unsigned short P[16 * 32];

    Frag qf[3];
    const unsigned short* qrow = qr + ((size_t)head * SEQ + q0 + l15) * HDP;
#pragma unroll
    for (int cc = 0; cc < 3; ++cc)
#pragma unroll
        for (int j = 0; j < 8; ++j) {
            int kk = cc * 32 + ((j >= 4) ? 16 : 0) + half * 8 + (j & 3) * 2;
            qf[cc].u[j] = *(const unsigned int*)(qrow + kk);
        }

    Acc o[5];
#pragma unroll
    for (int nb = 0; nb < 5; ++nb)
#pragma unroll
        for (int r = 0; r < 8; ++r) o[nb].f[r] = 0.f;
    float mr[8], lr[8];
#pragma unroll
    for (int r = 0; r < 8; ++r) { mr[r] = -3.0e38f; lr[r] = 0.f; }
    const float scale = 0.1118033989f;   // 1/sqrt(80)

    for (int n0 = 0; n0 < SEQ; n0 += 32) {
        Acc s0, s1;
#pragma unroll
        for (int r = 0; r < 8; ++r) { s0.f[r] = 0.f; s1.f[r] = 0.f; }
        const unsigned short* k0row = kr + ((size_t)head * SEQ + n0 + l15) * HDP;
        const unsigned short* k1row = k0row + 16 * HDP;
#pragma unroll
        for (int cc = 0; cc < 3; ++cc) {
            Frag b0, b1;
#pragma unroll
            for (int j = 0; j < 8; ++j) {
                int off = cc * 32 + ((j >= 4) ? 16 : 0) + half * 8 + (j & 3) * 2;
                b0.u[j] = *(const unsigned int*)(k0row + off);
                b1.u[j] = *(const unsigned int*)(k1row + off);
            }
            s0.v = __builtin_amdgcn_wmma_f32_16x16x32_bf16(
                false, qf[cc].v, false, b0.v, (short)0, s0.v, false, false);
            s1.v = __builtin_amdgcn_wmma_f32_16x16x32_bf16(
                false, qf[cc].v, false, b1.v, (short)0, s1.v, false, false);
        }

#pragma unroll
        for (int r = 0; r < 8; ++r) {
            float a0 = s0.f[r] * scale;
            float a1 = s1.f[r] * scale;
            float mx = fmaxf(a0, a1);
#pragma unroll
            for (int m = 8; m >= 1; m >>= 1) mx = fmaxf(mx, __shfl_xor(mx, m, 32));
            float mnew = fmaxf(mr[r], mx);
            float p0 = expf(a0 - mnew);
            float p1 = expf(a1 - mnew);
            float ps = p0 + p1;
#pragma unroll
            for (int m = 8; m >= 1; m >>= 1) ps += __shfl_xor(ps, m, 32);
            float corr = expf(mr[r] - mnew);
            lr[r] = lr[r] * corr + ps;
            mr[r] = mnew;
#pragma unroll
            for (int nb = 0; nb < 5; ++nb) o[nb].f[r] *= corr;
            P[(r + half * 8) * 32 + l15]      = f2bf(p0);
            P[(r + half * 8) * 32 + 16 + l15] = f2bf(p1);
        }
        __syncthreads();

        Frag pf;
#pragma unroll
        for (int j = 0; j < 8; ++j) {
            int kk = ((j >= 4) ? 16 : 0) + half * 8 + (j & 3) * 2;
            pf.u[j] = *(const unsigned int*)&P[l15 * 32 + kk];
        }
#pragma unroll
        for (int nb = 0; nb < 5; ++nb) {
            Frag vf;
            const unsigned short* vrow =
                vt + ((size_t)head * HD + nb * 16 + l15) * SEQ + n0 + half * 8;
#pragma unroll
            for (int j = 0; j < 8; ++j)
                vf.u[j] = *(const unsigned int*)(vrow + ((j >= 4) ? 16 : 0) + (j & 3) * 2);
            o[nb].v = __builtin_amdgcn_wmma_f32_16x16x32_bf16(
                false, pf.v, false, vf.v, (short)0, o[nb].v, false, false);
        }
        __syncthreads();
    }

#pragma unroll
    for (int nb = 0; nb < 5; ++nb)
#pragma unroll
        for (int r = 0; r < 8; ++r) {
            int row = q0 + r + half * 8;
            float val = o[nb].f[r] / lr[r];
            obf[(size_t)row * EMBED + head * HD + nb * 16 + l15] = f2bf(val);
        }
}

// ---------------------------------------------------------------------------
__global__ __launch_bounds__(256)
void f32_to_bf16(const float* __restrict__ in, unsigned short* __restrict__ out, int n)
{
    int i = blockIdx.x * 256 + threadIdx.x;
    if (i < n) out[i] = f2bf(in[i]);
}

// ---------------------------------------------------------------------------
extern "C" void kernel_launch(void* const* d_in, const int* in_sizes, int n_in,
                              void* d_out, int out_size, void* d_ws, size_t ws_size,
                              hipStream_t stream)
{
    (void)in_sizes; (void)n_in; (void)out_size; (void)ws_size;

    const float* x      = (const float*)d_in[0];
    const float* patchw = (const float*)d_in[1];
    const float* ln1g   = (const float*)d_in[2];
    const float* ln1b   = (const float*)d_in[3];
    const float* qkvw   = (const float*)d_in[4];
    const float* qkvb   = (const float*)d_in[5];
    const float* projw  = (const float*)d_in[6];
    const float* projb  = (const float*)d_in[7];
    const float* ln2g   = (const float*)d_in[8];
    const float* ln2b   = (const float*)d_in[9];
    const float* fc1w   = (const float*)d_in[10];
    const float* fc1b   = (const float*)d_in[11];
    const float* fc2w   = (const float*)d_in[12];
    const float* fc2b   = (const float*)d_in[13];
    const float* mlng   = (const float*)d_in[14];
    const float* mlnb   = (const float*)d_in[15];
    const float* mfc1w  = (const float*)d_in[16];
    const float* mfc1b  = (const float*)d_in[17];
    const float* mfc2w  = (const float*)d_in[18];
    const float* mfc2b  = (const float*)d_in[19];

    char* ws = (char*)d_ws;
    size_t off = 0;
    auto alloc = [&](size_t bytes) -> void* {
        void* p = ws + off;
        off += (bytes + 255) & ~(size_t)255;
        return p;
    };
    float*          h    = (float*)alloc((size_t)SEQ * EMBED * 4);
    unsigned short* xbf  = (unsigned short*)alloc((size_t)SEQ * PATCH * 2);
    unsigned short* ybf  = (unsigned short*)alloc((size_t)SEQ * EMBED * 2);
    float*          qkvf = (float*)alloc((size_t)SEQ * 3 * EMBED * 4);
    unsigned short* qrb  = (unsigned short*)alloc((size_t)HEADS * SEQ * HDP * 2);
    unsigned short* krb  = (unsigned short*)alloc((size_t)HEADS * SEQ * HDP * 2);
    unsigned short* vtb  = (unsigned short*)alloc((size_t)HEADS * HD * SEQ * 2);
    unsigned short* obf  = (unsigned short*)alloc((size_t)SEQ * EMBED * 2);
    unsigned short* a1   = (unsigned short*)alloc((size_t)SEQ * MLPDIM * 2);

    // ---- patch embed: h = x @ patch_w ----
    f32_to_bf16<<<(SEQ * PATCH + 255) / 256, 256, 0, stream>>>(x, xbf, SEQ * PATCH);
    gemm_wmma<0, false><<<dim3(EMBED / 128, SEQ / 128), 256, 0, stream>>>(
        xbf, PATCH, patchw, EMBED, nullptr, nullptr, h, EMBED, SEQ, EMBED, PATCH);

    // ---- 8 transformer blocks ----
    for (int l = 0; l < DEPTH; ++l) {
        layernorm_bf16<<<SEQ, 256, 0, stream>>>(h, ln1g + l * EMBED, ln1b + l * EMBED, ybf, EMBED);
        gemm_wmma<0, false><<<dim3(3 * EMBED / 128, SEQ / 128), 256, 0, stream>>>(
            ybf, EMBED, qkvw + (size_t)l * EMBED * 3 * EMBED, 3 * EMBED,
            qkvb + (size_t)l * 3 * EMBED, nullptr, qkvf, 3 * EMBED, SEQ, 3 * EMBED, EMBED);
        rope_pack<<<SEQ, 256, 0, stream>>>(qkvf, qrb, krb, vtb);
        attn_wmma<<<HEADS * (SEQ / 16), 32, 0, stream>>>(qrb, krb, vtb, obf);
        gemm_wmma<1, false><<<dim3(EMBED / 128, SEQ / 128), 256, 0, stream>>>(
            obf, EMBED, projw + (size_t)l * EMBED * EMBED, EMBED,
            projb + (size_t)l * EMBED, h, h, EMBED, SEQ, EMBED, EMBED);
        layernorm_bf16<<<SEQ, 256, 0, stream>>>(h, ln2g + l * EMBED, ln2b + l * EMBED, ybf, EMBED);
        gemm_wmma<2, true><<<dim3(MLPDIM / 128, SEQ / 128), 256, 0, stream>>>(
            ybf, EMBED, fc1w + (size_t)l * EMBED * MLPDIM, MLPDIM,
            fc1b + (size_t)l * MLPDIM, nullptr, a1, MLPDIM, SEQ, MLPDIM, EMBED);
        gemm_wmma<1, false><<<dim3(EMBED / 128, SEQ / 128), 256, 0, stream>>>(
            a1, MLPDIM, fc2w + (size_t)l * MLPDIM * EMBED, EMBED,
            fc2b + (size_t)l * EMBED, h, h, EMBED, SEQ, EMBED, MLPDIM);
    }

    // ---- merger: LN -> (256 x 5120) -> GELU MLP -> (256 x 3584) ----
    layernorm_bf16<<<SEQ, 256, 0, stream>>>(h, mlng, mlnb, ybf, EMBED);
    gemm_wmma<3, true><<<dim3(MLPDIM / 128, (SEQ / 4) / 128), 256, 0, stream>>>(
        ybf, MLPDIM, mfc1w, MLPDIM, mfc1b, nullptr, a1, MLPDIM, SEQ / 4, MLPDIM, MLPDIM);
    gemm_wmma<0, false><<<dim3(HIDDEN / 128, (SEQ / 4) / 128), 256, 0, stream>>>(
        a1, MLPDIM, mfc2w, HIDDEN, mfc2b, nullptr, d_out, HIDDEN, SEQ / 4, HIDDEN, MLPDIM);
}
